// PosteriorRotationN_64854006170325
// MI455X (gfx1250) — matrix operand order repaired and checked by reading, not compiled, not run on hardware
//
#include <hip/hip_runtime.h>
#include <hip/hip_bf16.h>
#include <math.h>

#define BB   8
#define TT   120
#define NN   24
#define DD   48
#define CC   1128
#define DE   256
#define CPAD 1136   // 71 tiles of 16
#define DPAD 64     // K padded for 2x k=32 WMMA
#define EPS  0.1f

typedef __attribute__((ext_vector_type(16))) __bf16 v16bf;
typedef __attribute__((ext_vector_type(8)))  float  v8f;
typedef __attribute__((ext_vector_type(4)))  float  f4;

__device__ __forceinline__ unsigned short f2bf(float f) {
  unsigned int u = __float_as_uint(f);
  unsigned int r = u + 0x7FFFu + ((u >> 16) & 1u);
  return (unsigned short)(r >> 16);
}

// Packed f32x2 -> bf16x2 (RNE). Prefer the single-instruction
// V_CVT_PK_BF16_F32 if the toolchain exposes it for gfx1250.
#if __has_builtin(__builtin_amdgcn_cvt_pk_bf16_f32)
typedef __attribute__((ext_vector_type(2))) __bf16 v2bf;
__device__ __forceinline__ unsigned int pack2(float a, float b) {
  union { v2bf v; unsigned int u; } cv;
  cv.v = __builtin_amdgcn_cvt_pk_bf16_f32(a, b);
  return cv.u;
}
#else
__device__ __forceinline__ unsigned int pack2(float a, float b) {
  return (unsigned int)f2bf(a) | ((unsigned int)f2bf(b) << 16);
}
#endif

union Frag { unsigned int u[8]; v16bf v; };

// ---------------------------------------------------------------------------
// Kernel 0: Wkv f32 [48][256] -> bf16 [DPAD=64][256] (rows 48..63 zero)
// ---------------------------------------------------------------------------
__global__ void wkv_prep_kernel(const float* __restrict__ Wkv,
                                unsigned short* __restrict__ wbf) {
  int idx = blockIdx.x * blockDim.x + threadIdx.x;
  if (idx >= DPAD * DE) return;
  int d = idx / DE, e = idx % DE;
  float f = (d < DD) ? Wkv[d * DE + e] : 0.0f;
  wbf[idx] = f2bf(f);
}

// ---------------------------------------------------------------------------
// Kernel 1: kv[b,c,n,d] = sum_e ctx[b,c,n,e]*Wkv[d,e] + bkv[d]
// Output workspace layout: kvbf[b][n][c(CPAD)][d(DPAD)] bf16
// One wave per (b, n, 16-row c-tile).  24 bf16 WMMAs per wave.
// Context loads are non-temporal (one-shot 222MB stream; keep L2 for kvbf)
// and software-pipelined one K-chunk ahead, with a global_prefetch further out.
// ---------------------------------------------------------------------------
__global__ void kv_gemm_kernel(const float* __restrict__ ctx,
                               const float* __restrict__ bkv,
                               const unsigned short* __restrict__ wbf,
                               unsigned short* __restrict__ kvbf) {
  const int ct   = blockIdx.x;   // 0..70
  const int n    = blockIdx.y;
  const int b    = blockIdx.z;
  const int lane = threadIdx.x;  // 0..31
  const int hi   = lane >> 4;
  const int lm   = lane & 15;

  const int  c0    = ct * 16;
  const int  crow  = c0 + lm;                 // A-operand row M = lane%16
  const bool cval  = (crow < CC);
  const float* arow = ctx + (((size_t)b * CC + crow) * NN + n) * DE;

  v8f acc[3] = {};                            // 3 d-tiles of 16 cols

  // Pipelined A: raw f32 chunks for current ks (zeros if row invalid —
  // pack2(0,0)==0, so the pack path below stays EXEC-uniform).
  f4 rf0 = {}, rf1 = {}, rg0 = {}, rg1 = {};
  if (cval) {
    const float* p = arow + hi * 16;
    rf0 = __builtin_nontemporal_load((const f4*)(p));
    rf1 = __builtin_nontemporal_load((const f4*)(p + 4));
    rg0 = __builtin_nontemporal_load((const f4*)(p + 8));
    rg1 = __builtin_nontemporal_load((const f4*)(p + 12));
    __builtin_prefetch(p + 32, 0, 0);          // global_prefetch one chunk ahead
  }

  for (int ks = 0; ks < 8; ++ks) {            // K = 256 in chunks of 32
    Frag A;                                    // pack current (uniform EXEC)
    A.u[0] = pack2(rf0.x, rf0.y); A.u[1] = pack2(rf0.z, rf0.w);
    A.u[2] = pack2(rf1.x, rf1.y); A.u[3] = pack2(rf1.z, rf1.w);
    A.u[4] = pack2(rg0.x, rg0.y); A.u[5] = pack2(rg0.z, rg0.w);
    A.u[6] = pack2(rg1.x, rg1.y); A.u[7] = pack2(rg1.z, rg1.w);

    if (ks < 7 && cval) {                      // issue next-ks loads now;
      const float* p = arow + (ks + 1) * 32 + hi * 16;  // wait lands next iter
      rf0 = __builtin_nontemporal_load((const f4*)(p));
      rf1 = __builtin_nontemporal_load((const f4*)(p + 4));
      rg0 = __builtin_nontemporal_load((const f4*)(p + 8));
      rg1 = __builtin_nontemporal_load((const f4*)(p + 12));
      if (ks < 6) __builtin_prefetch(p + 32, 0, 0);
    }

    #pragma unroll
    for (int dt = 0; dt < 3; ++dt) {
      Frag Bf;                                 // B[k=e][n=d] = Wkv[d][e]
      const int dcol  = dt * 16 + lm;          // B-operand col N = lane%16
      const int ebase = ks * 32 + hi * 16;     // K pairs 2v + 16*hi + 32*ks
      const unsigned int* wrow =
          (const unsigned int*)(wbf + (size_t)dcol * DE + ebase);
      #pragma unroll
      for (int v = 0; v < 8; ++v) Bf.u[v] = wrow[v];
      acc[dt] = __builtin_amdgcn_wmma_f32_16x16x32_bf16(
          false, A.v, false, Bf.v, (short)0, acc[dt], false, false);
    }
  }

  // Store bf16 to kvbf[b][n][c][d]; rows c>=CC forced to zero.
  const size_t base = (((size_t)b * NN + n) * CPAD) * DPAD;
  #pragma unroll
  for (int dt = 0; dt < 3; ++dt) {
    const float bias = bkv[dt * 16 + lm];
    #pragma unroll
    for (int v = 0; v < 8; ++v) {
      const int c = c0 + v + 8 * hi;           // C/D layout: M = v + 8*hi
      float val = (c < CC) ? (acc[dt][v] + bias) : 0.0f;
      kvbf[base + (size_t)c * DPAD + dt * 16 + lm] = f2bf(val);
    }
  }
  if (hi == 0) {                               // zero the d=48..63 K-padding
    const int d = 48 + lm;
    #pragma unroll
    for (int v = 0; v < 16; ++v)
      kvbf[base + (size_t)(c0 + v) * DPAD + d] = 0;
  }
}

// ---------------------------------------------------------------------------
// Kernel 2: per (b, n, 16-row t-tile):
//   sp[t,c] = sum_d x[t,d]*kv[c,d]  (bf16 WMMA, K=64 padded)
//   s = EPS*tanh(sp); pair (i,j) of c: dx[t,j]+=x[t,i]*s; dx[t,i]-=x[t,j]*s
//   y = x + dx   (dx accumulated in LDS via ds_add_f32)
// Block = 128 threads (4 waves); c-tiles strided across waves.
// B fragments double-buffered: next tile's loads issue before the current
// tile's tanh + ds_add_f32 scatter, which hides the (L2-resident) latency.
// ---------------------------------------------------------------------------
__global__ void rot_apply_kernel(const float* __restrict__ x,
                                 const unsigned short* __restrict__ kvbf,
                                 float* __restrict__ out) {
  const int tt   = blockIdx.x;   // 0..7 (T padded to 128)
  const int n    = blockIdx.y;
  const int b    = blockIdx.z;
  const int tid  = threadIdx.x;  // 0..127
  const int lane = tid & 31;
  const int wave = tid >> 5;
  const int hi   = lane >> 4;
  const int lm   = lane & 15;

  __shared__ float        xf[16][DD];   // x tile, f32 (for scatter operands)
  __shared__ unsigned int xbp[16][32];  // x tile, bf16 pairs, d padded to 64
  __shared__ float        dxs[16][DD];  // y accumulator (init = x)

  for (int idx = tid; idx < 16 * DD; idx += 128) {
    const int t = idx / DD, d = idx % DD;
    const int gt = tt * 16 + t;
    float v = (gt < TT) ? x[(((size_t)b * TT + gt) * NN + n) * DD + d] : 0.0f;
    xf[t][d]  = v;
    dxs[t][d] = v;
  }
  for (int idx = tid; idx < 16 * 32; idx += 128) {
    const int t = idx >> 5, p = idx & 31;
    const int gt = tt * 16 + t;
    const int d0 = 2 * p;
    float a = 0.0f, c2 = 0.0f;
    if (gt < TT && d0 < DD)
      a = x[(((size_t)b * TT + gt) * NN + n) * DD + d0];
    if (gt < TT && d0 + 1 < DD)
      c2 = x[(((size_t)b * TT + gt) * NN + n) * DD + d0 + 1];
    xbp[t][p] = pack2(a, c2);
  }
  __syncthreads();

  // Hoisted A fragments (same for every c-tile): K = 0..31 and 32..63
  Frag A0, A1;
  #pragma unroll
  for (int i = 0; i < 4; ++i) {
    const int d = hi * 16 + 2 * i;             // A K set {16hi+0..15}
    A0.u[i]     = xbp[lm][d >> 1];
    A0.u[i + 4] = xbp[lm][(d + 8) >> 1];
    A1.u[i]     = xbp[lm][(d + 32) >> 1];
    A1.u[i + 4] = xbp[lm][(d + 40) >> 1];
  }

  const size_t kvbase = (((size_t)b * NN + n) * CPAD) * DPAD;
  const int NTILE = CPAD / 16;                 // 71

  // Preload first B tile for this wave.
  Frag B0c = {}, B1c = {};
  {
    const int ccol = wave * 16 + lm;
    const unsigned int* brow =
        (const unsigned int*)(kvbf + kvbase + (size_t)ccol * DPAD + hi * 16);
    #pragma unroll
    for (int v = 0; v < 8; ++v) { B0c.u[v] = brow[v]; B1c.u[v] = brow[v + 16]; }
  }

  for (int ctIt = wave; ctIt < NTILE; ctIt += 4) {   // wave-uniform loop
    // Issue next tile's B loads before consuming the current one.
    Frag B0n = {}, B1n = {};
    const int ctn = ctIt + 4;
    if (ctn < NTILE) {
      const int ccoln = ctn * 16 + lm;
      const unsigned int* brow =
          (const unsigned int*)(kvbf + kvbase + (size_t)ccoln * DPAD + hi * 16);
      #pragma unroll
      for (int v = 0; v < 8; ++v) { B0n.u[v] = brow[v]; B1n.u[v] = brow[v + 16]; }
    }

    v8f acc = {};
    acc = __builtin_amdgcn_wmma_f32_16x16x32_bf16(
        false, A0.v, false, B0c.v, (short)0, acc, false, false);
    acc = __builtin_amdgcn_wmma_f32_16x16x32_bf16(
        false, A1.v, false, B1c.v, (short)0, acc, false, false);

    const int c = ctIt * 16 + lm;
    if (c < CC) {
      // c = i*(i-1)/2 + j  (lower triangle, i>j), recover (i,j)
      int i = (int)floorf((1.0f + sqrtf(1.0f + 8.0f * (float)c)) * 0.5f);
      while (i * (i - 1) / 2 > c) --i;
      while ((i + 1) * i / 2 <= c) ++i;
      const int j = c - i * (i - 1) / 2;
      #pragma unroll
      for (int v = 0; v < 8; ++v) {
        const int t = v + 8 * hi;                  // C/D layout: M = v + 8*hi
        const float s  = EPS * tanhf(acc[v]);
        const float xi = xf[t][i];
        const float xj = xf[t][j];
        atomicAdd(&dxs[t][j],  xi * s);            // ds_add_f32
        atomicAdd(&dxs[t][i], -xj * s);
      }
    }
    B0c = B0n; B1c = B1n;
  }
  __syncthreads();

  for (int idx = tid; idx < 16 * DD; idx += 128) {
    const int t = idx / DD, d = idx % DD;
    const int gt = tt * 16 + t;
    if (gt < TT)
      out[(((size_t)b * TT + gt) * NN + n) * DD + d] = dxs[t][d];
  }
}

// ---------------------------------------------------------------------------
extern "C" void kernel_launch(void* const* d_in, const int* in_sizes, int n_in,
                              void* d_out, int out_size, void* d_ws, size_t ws_size,
                              hipStream_t stream) {
  const float* x    = (const float*)d_in[0];
  const float* ctx  = (const float*)d_in[1];
  const float* Wkv  = (const float*)d_in[2];
  const float* bkv  = (const float*)d_in[3];
  float*       out  = (float*)d_out;

  unsigned short* kvbf = (unsigned short*)d_ws;                 // ~27.9 MB
  const size_t kvElems = (size_t)BB * NN * CPAD * DPAD;
  unsigned short* wbf  = kvbf + kvElems;                        // 32 KB

  wkv_prep_kernel<<<(DPAD * DE + 255) / 256, 256, 0, stream>>>(Wkv, wbf);

  dim3 g1(CPAD / 16, NN, BB);          // 71 x 24 x 8
  kv_gemm_kernel<<<g1, 32, 0, stream>>>(ctx, bkv, wbf, kvbf);

  rot_apply_kernel<<<dim3(8, NN, BB), 128, 0, stream>>>(x, kvbf, out);
}
